// TieredModel_35270271435217
// MI455X (gfx1250) — matrix operand order repaired and checked by reading, not compiled, not run on hardware
//
#include <hip/hip_runtime.h>
#include <hip/hip_bf16.h>

#define V_      32000
#define D_      512
#define B_      32
#define T_      256
#define FAST_   32
#define SLOW_   96
#define NSLOT_  128
#define ROWS_   (B_ * T_)      /* 8192 */
#define TSTEPS_ (T_ - 3)       /* 253  */

typedef __attribute__((ext_vector_type(16))) _Float16 v16h;
typedef __attribute__((ext_vector_type(8)))  float    v8f;

// ---------------------------------------------------------------------------
// Embedding gather: h0_f32[r,:] = embed[seq[r],:], plus f16 copy for WMMA.
// ---------------------------------------------------------------------------
__global__ __launch_bounds__(256)
void gather_embed(const int* __restrict__ seq, const float* __restrict__ embed,
                  float* __restrict__ h0f, _Float16* __restrict__ h0h)
{
    const int r   = blockIdx.x;                 // 0..8191
    const int tok = seq[r];
    const float* src = embed + (size_t)tok * D_;
    for (int i = threadIdx.x; i < D_; i += 256) {
        float v = src[i];
        h0f[(size_t)r * D_ + i] = v;
        h0h[(size_t)r * D_ + i] = (_Float16)v;
    }
}

// ---------------------------------------------------------------------------
// f32 -> f16 conversion (grid-stride).
// ---------------------------------------------------------------------------
__global__ __launch_bounds__(256)
void cvt16(const float* __restrict__ s, _Float16* __restrict__ d, int n)
{
    for (int i = blockIdx.x * 256 + threadIdx.x; i < n; i += gridDim.x * 256)
        d[i] = (_Float16)s[i];
}

// ---------------------------------------------------------------------------
// Tiled WMMA GEMM:  C[M,N] = A[M,K] * B[K,N] + bias,  f16 inputs, f32 accum.
//
// Block tile 128x128, 256 threads = 8 waves arranged 4x2; each wave owns a
// 32x64 strip: 2 A-fragments x 4 B-fragments -> 8 v_wmma_f32_16x16x32_f16
// per 32-wide K-step.
//
// Staging uses CDNA5 async copies: global_load_async_to_lds_b128 (ASYNCcnt,
// ISA 15.18.3 op 98) moves 16B per lane directly into LDS, then
// s_wait_asynccnt 0 + workgroup barrier. LDS tiles are unpadded so every
// 16-byte chunk is 16B-aligned (A row stride 64B, B row stride 256B).
//
// Fragment layouts per CDNA5 ISA 7.12.2:
//   A 16x32 f16 : lanes 0-15 -> M=0..15 (K 0..7,16..23), lanes 16-31 same M
//                 with K 8..15,24..31
//   B 32x16 f16 : N = lane&15, 16 f16 = K rows in the same K order as A
//   C/D 16x16   : v8f, n = lane&15, m = r + 8*(lane>>4)
// Requires: K % 32 == 0, N % 128 == 0 (true for all call sites here).
// Out-of-range A rows are clamped to M-1 (results discarded by epilogue).
// ---------------------------------------------------------------------------
template<int RELU, int STORE32, int STORE16>
__global__ __launch_bounds__(256)
void wmma_gemm(const _Float16* __restrict__ A, const _Float16* __restrict__ Bm,
               const float* __restrict__ bias,
               float* __restrict__ C32, _Float16* __restrict__ C16,
               int M, int N, int K)
{
    __shared__ _Float16 As[128][32];   // 8 KB, row stride 64 B
    __shared__ _Float16 Bs[32][128];   // 8 KB, row stride 256 B

    const int tid    = threadIdx.x;
    const int wid    = tid >> 5;
    const int lane   = tid & 31;
    const int blockM = blockIdx.y * 128;
    const int blockN = blockIdx.x * 128;

    const int waveM = (wid >> 1) * 32;  // 0,32,64,96
    const int waveN = (wid & 1) * 64;   // 0,64

    v8f acc[2][4] = {};

    for (int k0 = 0; k0 < K; k0 += 32) {
        // ---- async stage A tile 128x32 (2 x 16B chunks per thread) ----
        #pragma unroll
        for (int c = 0; c < 2; ++c) {
            const int idx = tid + c * 256;
            const int row = idx >> 2;            // 0..127
            const int col = (idx & 3) * 8;       // 0,8,16,24
            int gr = blockM + row;
            if (gr >= M) gr = M - 1;             // clamp: discarded later
            const unsigned long long gp =
                (unsigned long long)(uintptr_t)(A + (size_t)gr * K + k0 + col);
            const unsigned lo = (unsigned)(uintptr_t)(&As[row][col]);
            asm volatile("global_load_async_to_lds_b128 %0, %1, off"
                         :: "v"(lo), "v"(gp) : "memory");
        }
        // ---- async stage B tile 32x128 ----
        #pragma unroll
        for (int c = 0; c < 2; ++c) {
            const int idx = tid + c * 256;
            const int row = idx >> 4;            // 0..31
            const int col = (idx & 15) * 8;      // 0..120
            const unsigned long long gp =
                (unsigned long long)(uintptr_t)(Bm + (size_t)(k0 + row) * N + blockN + col);
            const unsigned lo = (unsigned)(uintptr_t)(&Bs[row][col]);
            asm volatile("global_load_async_to_lds_b128 %0, %1, off"
                         :: "v"(lo), "v"(gp) : "memory");
        }
        // prefetch next B tile into GL2: global_prefetch_b8
        if (k0 + 32 < K) {
            __builtin_prefetch(
                (const void*)(Bm + (size_t)(k0 + 32 + (tid >> 4)) * N + blockN + (tid & 15) * 8),
                0, 1);
        }
        asm volatile("s_wait_asynccnt 0x0" ::: "memory");
        __syncthreads();

        // ---- build fragments ----
        v16h a[2];
        {
            const int m  = lane & 15;
            const int kh = lane >> 4;
            #pragma unroll
            for (int i = 0; i < 2; ++i) {
                #pragma unroll
                for (int j = 0; j < 8; ++j) a[i][j]     = As[waveM + i * 16 + m][kh * 8 + j];
                #pragma unroll
                for (int j = 0; j < 8; ++j) a[i][8 + j] = As[waveM + i * 16 + m][16 + kh * 8 + j];
            }
        }
        v16h bfr[4];
        #pragma unroll
        for (int s = 0; s < 4; ++s) {
            #pragma unroll
            for (int j = 0; j < 16; ++j) bfr[s][j] = Bs[(j & 8) * 2 + (lane >> 4) * 8 + (j & 7)][waveN + s * 16 + (lane & 15)];
        }

        #pragma unroll
        for (int i = 0; i < 2; ++i) {
            #pragma unroll
            for (int s = 0; s < 4; ++s) {
                acc[i][s] = __builtin_amdgcn_wmma_f32_16x16x32_f16(
                    false, a[i], false, bfr[s], (short)0, acc[i][s], false, false);
            }
        }
        __syncthreads();
    }

    // ---- epilogue: bias (+ReLU), store f32 and/or f16 ----
    #pragma unroll
    for (int i = 0; i < 2; ++i) {
        #pragma unroll
        for (int s = 0; s < 4; ++s) {
            const int n  = blockN + waveN + s * 16 + (lane & 15);
            const int mb = blockM + waveM + i * 16 + ((lane >> 4) << 3);
            const float bs = bias[n];
            #pragma unroll
            for (int r = 0; r < 8; ++r) {
                const int m = mb + r;
                if (m < M) {
                    float v = acc[i][s][r] + bs;
                    if (RELU) v = v > 0.f ? v : 0.f;
                    if (STORE32) C32[(size_t)m * N + n] = v;
                    if (STORE16) C16[(size_t)m * N + n] = (_Float16)v;
                }
            }
        }
    }
}

// ---------------------------------------------------------------------------
// Residual + LayerNorm: h = LN(x_ff + h0) * g + b, one block per row (D=512).
// ---------------------------------------------------------------------------
__global__ __launch_bounds__(256)
void residual_ln(const float* __restrict__ xf, const float* __restrict__ h0,
                 const float* __restrict__ g, const float* __restrict__ bb,
                 float* __restrict__ h)
{
    __shared__ float red[256];
    const int r   = blockIdx.x;
    const int tid = threadIdx.x;
    const size_t base = (size_t)r * D_;

    float v0 = xf[base + tid]       + h0[base + tid];
    float v1 = xf[base + tid + 256] + h0[base + tid + 256];

    red[tid] = v0 + v1;
    __syncthreads();
    for (int s = 128; s > 0; s >>= 1) {
        if (tid < s) red[tid] += red[tid + s];
        __syncthreads();
    }
    const float mu = red[0] * (1.f / (float)D_);
    __syncthreads();

    const float d0 = v0 - mu, d1 = v1 - mu;
    red[tid] = d0 * d0 + d1 * d1;
    __syncthreads();
    for (int s = 128; s > 0; s >>= 1) {
        if (tid < s) red[tid] += red[tid + s];
        __syncthreads();
    }
    const float inv = rsqrtf(red[0] * (1.f / (float)D_) + 1e-5f);

    h[base + tid]       = d0 * inv * g[tid]       + bb[tid];
    h[base + tid + 256] = d1 * inv * g[tid + 256] + bb[tid + 256];
}

// ---------------------------------------------------------------------------
// Sequential tiered-memory scan. One workgroup per batch element.
// Fast memory (32x512 f32 = 64KB) lives in LDS; slow memory rows are written
// directly to the global `mem` buffer on demotes. Scalar slot logic in lane 0.
// ---------------------------------------------------------------------------
__global__ __launch_bounds__(256)
void scan_kernel(const float* __restrict__ h,
                 const float* __restrict__ wg_w, const float* __restrict__ wg_b,
                 const float* __restrict__ dem_w, const float* __restrict__ dem_b,
                 float* __restrict__ mem, float* __restrict__ mask)
{
    extern __shared__ float smem[];
    float* fm     = smem;                 // FAST_*D_
    float* tokbuf = fm + FAST_ * D_;      // D_
    float* ds     = tokbuf + D_;          // FAST_
    __shared__ float red[256];
    __shared__ int   fa[FAST_], sa[SLOW_];
    __shared__ unsigned char fu[FAST_], su[SLOW_];
    __shared__ int s_target, s_dem, s_ss, s_write, s_demote;

    const int b   = blockIdx.x;
    const int tid = threadIdx.x;
    float* memb = mem + (size_t)b * NSLOT_ * D_;

    for (int i = tid; i < FAST_ * D_; i += 256) fm[i] = 0.f;
    for (int i = tid; i < NSLOT_ * D_; i += 256) memb[i] = 0.f;
    if (tid < FAST_) { fa[tid] = 0; fu[tid] = 0; }
    if (tid < SLOW_) { sa[tid] = 0; su[tid] = 0; }
    __syncthreads();

    const float wgb  = wg_b[0];
    const float demb = dem_b[0];

    for (int t = 0; t < TSTEPS_; ++t) {
        const float* tok = h + ((size_t)b * T_ + t) * D_;
        // load token, partial dot with wg_w
        float p = 0.f;
        #pragma unroll
        for (int u = 0; u < 2; ++u) {
            int i = tid + u * 256;
            float v = tok[i];
            tokbuf[i] = v;
            p += v * wg_w[i];
        }
        red[tid] = p;
        __syncthreads();
        for (int s = 128; s > 0; s >>= 1) {
            if (tid < s) red[tid] += red[tid + s];
            __syncthreads();
        }

        // age increment (before any reset, matching the reference)
        if (tid < FAST_ && fu[tid]) fa[tid]++;
        if (tid < SLOW_ && su[tid]) sa[tid]++;

        // ds = fm @ dem_w + dem_b  (8 partial lanes per slot, LDS f32 atomics)
        if (tid < FAST_) ds[tid] = demb;
        __syncthreads();
        {
            const int slot = tid >> 3;
            const int part = tid & 7;
            const float* fr = fm + slot * D_;
            float acc = 0.f;
            for (int j = part * 64; j < part * 64 + 64; ++j)
                acc += fr[j] * dem_w[j];
            atomicAdd(&ds[slot], acc);
        }
        __syncthreads();

        if (tid == 0) {
            const float wsig = 1.f / (1.f + __expf(-(red[0] + wgb)));
            const int wr = (wsig >= 0.4f) ? 1 : 0;
            int ff = -1;
            for (int i = 0; i < FAST_; ++i) if (!fu[i]) { ff = i; break; }
            int dem = 0; float best = ds[0];
            for (int i = 1; i < FAST_; ++i)
                if (ds[i] < best) { best = ds[i]; dem = i; }
            const int has_free = (ff >= 0);
            s_dem    = dem;
            s_target = has_free ? ff : dem;
            s_write  = wr;
            const int demote = wr && !has_free;
            s_demote = demote;
            if (demote) {
                int fs = -1;
                for (int i = 0; i < SLOW_; ++i) if (!su[i]) { fs = i; break; }
                int ss;
                if (fs >= 0) ss = fs;
                else {
                    int bi = 0, bv = sa[0];
                    for (int i = 1; i < SLOW_; ++i)
                        if (sa[i] > bv) { bv = sa[i]; bi = i; }
                    ss = bi;
                }
                s_ss = ss;
            }
        }
        __syncthreads();

        if (s_demote) {
            float* dst = memb + (size_t)(FAST_ + s_ss) * D_;
            const float* srcr = fm + s_dem * D_;
            for (int i = tid; i < D_; i += 256) dst[i] = srcr[i];
            if (tid == 0) { sa[s_ss] = 0; su[s_ss] = 1; }
        }
        __syncthreads();   // demoted row copied out before fm overwrite

        if (s_write) {
            float* dst = fm + s_target * D_;
            for (int i = tid; i < D_; i += 256) dst[i] = tokbuf[i];
            if (tid == 0) { fa[s_target] = 0; fu[s_target] = 1; }
        }
        __syncthreads();
    }

    // write back fast memory + usage mask
    for (int i = tid; i < FAST_ * D_; i += 256) memb[i] = fm[i];
    if (tid < FAST_) mask[b * NSLOT_ + tid]         = fu[tid] ? 1.f : 0.f;
    if (tid < SLOW_) mask[b * NSLOT_ + FAST_ + tid] = su[tid] ? 1.f : 0.f;
}

// ---------------------------------------------------------------------------
// Read head: q = h_last @ q_w + q_b; masked softmax attention over 128 slots;
// ctx written as f16 for the output WMMA GEMM. One block per batch element.
// ---------------------------------------------------------------------------
__global__ __launch_bounds__(256)
void readhead(const float* __restrict__ h,
              const float* __restrict__ q_w, const float* __restrict__ q_b,
              const float* __restrict__ mem, const float* __restrict__ mask,
              _Float16* __restrict__ ctx16)
{
    __shared__ float hl[D_];
    __shared__ float qv[D_];
    __shared__ float sc[NSLOT_];
    __shared__ float attn[NSLOT_];
    __shared__ float s_sum;

    const int b   = blockIdx.x;
    const int tid = threadIdx.x;
    const float* hlast = h + ((size_t)b * T_ + (T_ - 1)) * D_;
    const float* memb  = mem + (size_t)b * NSLOT_ * D_;

    for (int i = tid; i < D_; i += 256) hl[i] = hlast[i];
    __syncthreads();

    for (int j = tid; j < D_; j += 256) {
        float acc = q_b[j];
        for (int i = 0; i < D_; ++i) acc += hl[i] * q_w[(size_t)i * D_ + j];
        qv[j] = acc;
    }
    __syncthreads();

    if (tid < NSLOT_) {
        const float* mr = memb + (size_t)tid * D_;
        float acc = 0.f;
        for (int i = 0; i < D_; ++i) acc += mr[i] * qv[i];
        sc[tid] = (mask[b * NSLOT_ + tid] == 0.f) ? -1e9f : acc;
    }
    __syncthreads();

    if (tid == 0) {
        float mx = sc[0];
        for (int i = 1; i < NSLOT_; ++i) mx = fmaxf(mx, sc[i]);
        float sum = 0.f;
        for (int i = 0; i < NSLOT_; ++i) {
            float e = __expf(sc[i] - mx);
            attn[i] = e;
            sum += e;
        }
        s_sum = sum;
    }
    __syncthreads();

    const float inv = 1.f / s_sum;
    for (int d = tid; d < D_; d += 256) {
        float acc = 0.f;
        for (int n = 0; n < NSLOT_; ++n)
            acc += attn[n] * memb[(size_t)n * D_ + d];
        ctx16[(size_t)b * D_ + d] = (_Float16)(acc * inv);
    }
}

// ---------------------------------------------------------------------------
// Launcher
// ---------------------------------------------------------------------------
extern "C" void kernel_launch(void* const* d_in, const int* in_sizes, int n_in,
                              void* d_out, int out_size, void* d_ws, size_t ws_size,
                              hipStream_t stream)
{
    (void)in_sizes; (void)n_in; (void)out_size; (void)ws_size;

    const int*   seq   = (const int*)  d_in[0];
    const float* embed = (const float*)d_in[1];
    const float* ff1_w = (const float*)d_in[2];
    const float* ff1_b = (const float*)d_in[3];
    const float* ff2_w = (const float*)d_in[4];
    const float* ff2_b = (const float*)d_in[5];
    const float* ln_g  = (const float*)d_in[6];
    const float* ln_b  = (const float*)d_in[7];
    const float* wg_w  = (const float*)d_in[8];
    const float* wg_b  = (const float*)d_in[9];
    const float* dem_w = (const float*)d_in[10];
    const float* dem_b = (const float*)d_in[11];
    const float* q_w   = (const float*)d_in[12];
    const float* q_b   = (const float*)d_in[13];
    const float* out_w = (const float*)d_in[14];
    const float* out_b = (const float*)d_in[15];
    float* out = (float*)d_out;

    char* ws = (char*)d_ws;
    size_t off = 0;
    auto alloc = [&](size_t bytes) -> char* {
        char* p = ws + off;
        off = (off + bytes + 255) & ~(size_t)255;
        return p;
    };
    float*    h0f   = (float*)   alloc((size_t)ROWS_ * D_ * 4);
    _Float16* h0h   = (_Float16*)alloc((size_t)ROWS_ * D_ * 2);
    _Float16* w1h   = (_Float16*)alloc((size_t)D_ * 2 * D_ * 2);
    _Float16* w2h   = (_Float16*)alloc((size_t)2 * D_ * D_ * 2);
    _Float16* wouth = (_Float16*)alloc((size_t)D_ * V_ * 2);
    _Float16* ff1o  = (_Float16*)alloc((size_t)ROWS_ * 2 * D_ * 2);
    float*    xf    = (float*)   alloc((size_t)ROWS_ * D_ * 4);
    float*    hbuf  = (float*)   alloc((size_t)ROWS_ * D_ * 4);
    float*    memb  = (float*)   alloc((size_t)B_ * NSLOT_ * D_ * 4);
    float*    maskb = (float*)   alloc((size_t)B_ * NSLOT_ * 4);
    _Float16* ctx16 = (_Float16*)alloc((size_t)B_ * D_ * 2);

    // 1. gather + f16 conversions
    gather_embed<<<ROWS_, 256, 0, stream>>>(seq, embed, h0f, h0h);
    cvt16<<<512, 256, 0, stream>>>(ff1_w, w1h, D_ * 2 * D_);
    cvt16<<<512, 256, 0, stream>>>(ff2_w, w2h, 2 * D_ * D_);
    cvt16<<<2048, 256, 0, stream>>>(out_w, wouth, D_ * V_);

    // 2. FFN GEMM 1: relu(h0 @ ff1_w + b) -> f16   [8192,1024]
    dim3 g1((2 * D_) / 128, ROWS_ / 128);
    wmma_gemm<1, 0, 1><<<g1, 256, 0, stream>>>(h0h, w1h, ff1_b, (float*)nullptr,
                                               ff1o, ROWS_, 2 * D_, D_);
    // 3. FFN GEMM 2: ff1o @ ff2_w + b -> f32       [8192,512]
    dim3 g2(D_ / 128, ROWS_ / 128);
    wmma_gemm<0, 1, 0><<<g2, 256, 0, stream>>>(ff1o, w2h, ff2_b, xf,
                                               (_Float16*)nullptr, ROWS_, D_, 2 * D_);
    // 4. residual + LayerNorm -> h
    residual_ln<<<ROWS_, 256, 0, stream>>>(xf, h0f, ln_g, ln_b, hbuf);

    // 5. tiered-memory scan (fast memory LDS-resident: 32*512*4 + tok + ds)
    size_t scan_lds = (size_t)(FAST_ * D_ + D_ + FAST_) * sizeof(float);
    scan_kernel<<<B_, 256, scan_lds, stream>>>(hbuf, wg_w, wg_b, dem_w, dem_b,
                                               memb, maskb);
    // 6. read head -> ctx (f16)
    readhead<<<B_, 256, 0, stream>>>(hbuf, q_w, q_b, memb, maskb, ctx16);

    // 7. output projection: ctx @ out_w + out_b -> logits [32,32000]
    dim3 g3(V_ / 128, 1);
    wmma_gemm<0, 1, 0><<<g3, 256, 0, stream>>>(ctx16, wouth, out_b, out,
                                               (_Float16*)nullptr, B_, V_, D_);
}
